// LSTMCell_19885698580623
// MI455X (gfx1250) — matrix operand order repaired and checked by reading, not compiled, not run on hardware
//
#include <hip/hip_runtime.h>
#include <hip/hip_bf16.h>

// Problem dims
#define BDIM 16384
#define INDIM 1024
#define HDIM 1024
#define KDIM 2048          // IN + H
#define KTILE 64           // staged K per iteration (2 WMMA K-steps)
#define KITERS (KDIM / KTILE)   // 32

typedef __attribute__((ext_vector_type(16))) __bf16 v16bf;
typedef __attribute__((ext_vector_type(8)))  float  v8f;

struct U32x8 { uint4 lo, hi; };          // 32 bytes == v16bf
union FragCast { U32x8 u; v16bf v; };

__device__ __forceinline__ unsigned short f2bf(float f) {
    unsigned int u = __float_as_uint(f);
    u += 0x7FFFu + ((u >> 16) & 1u);     // round-to-nearest-even
    return (unsigned short)(u >> 16);
}
__device__ __forceinline__ unsigned int pack2(float a, float b) {
    return (unsigned int)f2bf(a) | ((unsigned int)f2bf(b) << 16);
}
__device__ __forceinline__ float sigmoid_fast(float x) {
    return 1.0f / (1.0f + __expf(-x));
}

// ---- Pass 1a: A[b][k] = bf16( x[b][k] | h0[b][k-1024] ), 4 elems/thread ----
__global__ void lstm_cvt_act(const float* __restrict__ x, const float* __restrict__ h0,
                             unsigned short* __restrict__ A) {
    size_t e    = (size_t)blockIdx.x * blockDim.x + threadIdx.x;
    size_t idx4 = e * 4;                              // over B*IN elems
    size_t b    = idx4 >> 10;
    size_t i    = idx4 & 1023;
    float4 xv = *(const float4*)(x  + idx4);
    float4 hv = *(const float4*)(h0 + idx4);
    uint2 px, ph;
    px.x = pack2(xv.x, xv.y); px.y = pack2(xv.z, xv.w);
    ph.x = pack2(hv.x, hv.y); ph.y = pack2(hv.z, hv.w);
    size_t rowbase = b * (size_t)KDIM;
    *(uint2*)(A + rowbase + i)          = px;
    *(uint2*)(A + rowbase + INDIM + i)  = ph;
}

// ---- Pass 1b: W[g][h][k] = bf16( wi[g][h][k] | wh[g][h][k-1024] ) ----
__global__ void lstm_cvt_w(const float* __restrict__ wi, const float* __restrict__ wh,
                           unsigned short* __restrict__ W) {
    size_t e    = (size_t)blockIdx.x * blockDim.x + threadIdx.x;
    size_t idx4 = e * 4;                              // over 4*H*IN elems
    size_t gh   = idx4 >> 10;                         // g*H + h
    size_t k    = idx4 & 1023;
    float4 a = *(const float4*)(wi + idx4);
    float4 b = *(const float4*)(wh + idx4);
    uint2 pa, pb;
    pa.x = pack2(a.x, a.y); pa.y = pack2(a.z, a.w);
    pb.x = pack2(b.x, b.y); pb.y = pack2(b.z, b.w);
    size_t rowbase = gh * (size_t)KDIM;
    *(uint2*)(W + rowbase + k)          = pa;
    *(uint2*)(W + rowbase + INDIM + k)  = pb;
}

// ---- Pass 2: fused GEMM (bf16 WMMA, fp32 acc) + LSTM gate epilogue ----
// Workgroup tile: 128 M x 32 h x 4 gates. 8 waves split 4(M) x 2(N):
// each wave owns 32 M rows x 16 h cols x 4 gates = acc[4 gates][2 M-subtiles].
// Staged K-tile = 64: ALL fragments for the tile (4 A + 8 B) are loaded into
// distinct registers up front, then a 16-WMMA burst -> shallow dscnt waits.
__launch_bounds__(256)
__global__ void lstm_gemm(const unsigned short* __restrict__ A,
                          const unsigned short* __restrict__ W,
                          const float* __restrict__ bi, const float* __restrict__ bh,
                          const float* __restrict__ c0, float* __restrict__ out) {
    __shared__ __align__(16) unsigned short As[2][128 * KTILE];      // 2 x 16 KB
    __shared__ __align__(16) unsigned short Bs[2][4 * 32 * KTILE];   // 2 x 16 KB

    const int tid = threadIdx.x;
    const int m0  = blockIdx.y * 128;
    const int hb  = blockIdx.x * 32;

    // --- staging coordinates (each thread moves 64 B of A and 64 B of B) ---
    const int arow = tid >> 1;
    const int acol = (tid & 1) * 32;
    const unsigned short* gA = A + (size_t)(m0 + arow) * KDIM + acol;
    const int a_lds = arow * KTILE + acol;

    const int bg   = tid >> 6;            // gate 0..3
    const int brow = (tid >> 1) & 31;     // h row within tile
    const int bcol = (tid & 1) * 32;
    const unsigned short* gB = W + ((size_t)bg * HDIM + (hb + brow)) * KDIM + bcol;
    const int b_lds = (bg * 32 + brow) * KTILE + bcol;

    // prologue: stage k-tile 0 into buffer 0
    uint4 ra[4], rb[4];
    #pragma unroll
    for (int c = 0; c < 4; ++c) {
        ra[c] = *(const uint4*)(gA + c * 8);
        rb[c] = *(const uint4*)(gB + c * 8);
    }
    #pragma unroll
    for (int c = 0; c < 4; ++c) {
        *(uint4*)&As[0][a_lds + c * 8] = ra[c];
        *(uint4*)&Bs[0][b_lds + c * 8] = rb[c];
    }

    // --- per-wave WMMA coordinates: waves = 4 M-groups x 2 N-groups ---
    const int lane  = tid & 31;
    const int wv    = tid >> 5;
    const int mgrp  = wv >> 1;            // 0..3 -> M rows [mgrp*32, mgrp*32+32)
    const int ngrp  = wv & 1;             // 0..1 -> h cols [ngrp*16, ngrp*16+16)
    const int half  = lane >> 4;
    const int lr    = lane & 15;

    // A frag (16-bit A 16x32 layout): elems 0..7 at k=half*8, elems 8..15 at k=16+half*8
    const int a_frag0 = (mgrp * 32 + lr) * KTILE + half * 8;        // ms = 0
    const int a_frag1 = (mgrp * 32 + 16 + lr) * KTILE + half * 8;   // ms = 1
    // B frag (16-bit B 32x16 layout): lane column = ngrp*16+lr, elems 0..7 at
    // k=half*8, elems 8..15 at k=16+half*8; one frag per gate.
    const int b_frag  = (ngrp * 16 + lr) * KTILE + half * 8;        // + g*32*KTILE

    v8f acc[4][2] = {};                   // [gate][M-subtile]

    for (int t = 0; t < KITERS; ++t) {
        const int cur = t & 1;
        __syncthreads();
        if (t + 1 < KITERS) {             // fetch next k-tile into registers
            const unsigned short* pA = gA + (t + 1) * KTILE;
            const unsigned short* pB = gB + (t + 1) * KTILE;
            #pragma unroll
            for (int c = 0; c < 4; ++c) {
                ra[c] = *(const uint4*)(pA + c * 8);
                rb[c] = *(const uint4*)(pB + c * 8);
            }
        }
        if (t + 2 < KITERS) {             // prefetch two tiles ahead
            __builtin_prefetch((const void*)(gA + (t + 2) * KTILE), 0, 3);
            __builtin_prefetch((const void*)(gB + (t + 2) * KTILE), 0, 3);
        }

        // ---- load ALL fragments for this staged tile into distinct regs ----
        FragCast fb[2][4];                // [ks][gate]
        FragCast fa[2][2];                // [ks][M-subtile]
        #pragma unroll
        for (int ks = 0; ks < 2; ++ks) {
            const int kb = ks * 32;
            #pragma unroll
            for (int g = 0; g < 4; ++g) {
                const int boff = g * 32 * KTILE + b_frag + kb;
                fb[ks][g].u.lo = *(const uint4*)&Bs[cur][boff];
                fb[ks][g].u.hi = *(const uint4*)&Bs[cur][boff + 16];
            }
            fa[ks][0].u.lo = *(const uint4*)&As[cur][a_frag0 + kb];
            fa[ks][0].u.hi = *(const uint4*)&As[cur][a_frag0 + kb + 16];
            fa[ks][1].u.lo = *(const uint4*)&As[cur][a_frag1 + kb];
            fa[ks][1].u.hi = *(const uint4*)&As[cur][a_frag1 + kb + 16];
        }

        // ---- 16-WMMA burst ----
        #pragma unroll
        for (int ks = 0; ks < 2; ++ks) {
            #pragma unroll
            for (int g = 0; g < 4; ++g) {
                #pragma unroll
                for (int ms = 0; ms < 2; ++ms) {
                    acc[g][ms] = __builtin_amdgcn_wmma_f32_16x16x32_bf16(
                        false, fa[ks][ms].v, false, fb[ks][g].v, (short)0,
                        acc[g][ms], false, false);
                }
            }
        }

        if (t + 1 < KITERS) {             // stage next tile into the other buffer
            const int nxt = cur ^ 1;
            #pragma unroll
            for (int c = 0; c < 4; ++c) {
                *(uint4*)&As[nxt][a_lds + c * 8] = ra[c];
                *(uint4*)&Bs[nxt][b_lds + c * 8] = rb[c];
            }
        }
    }

    // --- LSTM epilogue, fully in-register (all 4 gates live in this lane) ---
    // C/D layout: lane (half,lr): col N = lr, row M = half*8 + v
    const size_t ct_base = (size_t)BDIM * HDIM;
    const int h = hb + ngrp * 16 + lr;
    float sb[4];
    #pragma unroll
    for (int g = 0; g < 4; ++g)
        sb[g] = bi[g * HDIM + h] + bh[g * HDIM + h];
    #pragma unroll
    for (int ms = 0; ms < 2; ++ms) {
        #pragma unroll
        for (int v = 0; v < 8; ++v) {
            const int row = m0 + mgrp * 32 + ms * 16 + half * 8 + v;
            const float zf = acc[0][ms][v] + sb[0];
            const float zi = acc[1][ms][v] + sb[1];
            const float zo = acc[2][ms][v] + sb[2];
            const float zg = acc[3][ms][v] + sb[3];
            const float f  = sigmoid_fast(zf);
            const float ii = sigmoid_fast(zi);
            const float o  = sigmoid_fast(zo);
            const float gg = tanhf(zg);
            const float cprev = c0[(size_t)row * HDIM + h];
            const float ct = f * cprev + ii * gg;
            const float ht = o * tanhf(ct);
            out[(size_t)row * HDIM + h]           = ht;
            out[ct_base + (size_t)row * HDIM + h] = ct;
        }
    }
}

extern "C" void kernel_launch(void* const* d_in, const int* in_sizes, int n_in,
                              void* d_out, int out_size, void* d_ws, size_t ws_size,
                              hipStream_t stream) {
    (void)in_sizes; (void)n_in; (void)out_size; (void)ws_size;
    const float* x  = (const float*)d_in[0];
    const float* h0 = (const float*)d_in[1];
    const float* c0 = (const float*)d_in[2];
    const float* wi = (const float*)d_in[3];
    const float* bi = (const float*)d_in[4];
    const float* wh = (const float*)d_in[5];
    const float* bh = (const float*)d_in[6];
    float* out = (float*)d_out;

    // Workspace layout: A bf16 [16384][2048] (64 MB) then W bf16 [4][1024][2048] (16 MB)
    unsigned short* Abf = (unsigned short*)d_ws;
    unsigned short* Wbf = (unsigned short*)((char*)d_ws + (size_t)BDIM * KDIM * sizeof(unsigned short));

    // Pass 1: fp32 -> bf16 conversion (operands then fit in 192 MB L2)
    lstm_cvt_act<<<(BDIM * INDIM / 4) / 256, 256, 0, stream>>>(x, h0, Abf);
    lstm_cvt_w<<<(4 * HDIM * INDIM / 4) / 256, 256, 0, stream>>>(wi, wh, Wbf);

    // Pass 2: fused dual-GEMM + gate epilogue
    dim3 grid(HDIM / 32, BDIM / 128);   // (32, 128)
    lstm_gemm<<<grid, 256, 0, stream>>>(Abf, Wbf, bi, bh, c0, out);
}
